// TransformerBasedModel_86852828659817
// MI455X (gfx1250) — matrix operand order repaired and checked by reading, not compiled, not run on hardware
//
#include <hip/hip_runtime.h>
#include <hip/hip_bf16.h>

// ---------------------------------------------------------------------------
// GNN TransformerConv (4 layers, H=4, C=128, HC=512) for MI455X / gfx1250.
// Dense linears via v_wmma_f32_16x16x32_f16 (f16 in, f32 acc); edge softmax
// via wave32 VALU + L2 atomics; two-pass batch norm; add-pool + small MLP.
// ---------------------------------------------------------------------------

typedef __attribute__((ext_vector_type(16))) _Float16 v16h;
typedef __attribute__((ext_vector_type(8)))  _Float16 v8h;
typedef __attribute__((ext_vector_type(8)))  float    v8f;

#define NN_NODES 10000
#define MP_NODES 10016            // padded to multiple of 16
#define N_EDGES  160000
#define N_GRAPHS 512
#define HC       512
#define K0_PAD   96               // 76 padded to multiple of 32

// ---------------- order-preserving float<->uint for atomicMax --------------
__device__ __forceinline__ unsigned fmap(float x) {
  unsigned u = __float_as_uint(x);
  return (u & 0x80000000u) ? ~u : (u | 0x80000000u);
}
__device__ __forceinline__ float funmap(unsigned u) {
  return (u & 0x80000000u) ? __uint_as_float(u & 0x7fffffffu)
                           : __uint_as_float(~u);
}
#define AMAX_NEG_INF 0x007FFFFFu   // fmap(-inf)

// ---------------------------------------------------------------------------
// Weight prep: W f32 [fin x 512] row-major  ->  Wt f16 [512 x Kp] (transposed,
// zero-padded K) so each WMMA B-fragment lane load is contiguous.
// ---------------------------------------------------------------------------
__global__ void wprep_kernel(const float* __restrict__ W, _Float16* __restrict__ Wt,
                             int fin, int Kp) {
  int idx = blockIdx.x * blockDim.x + threadIdx.x;
  if (idx >= 512 * Kp) return;
  int n = idx / Kp, k = idx - n * Kp;
  float v = (k < fin) ? W[(size_t)k * 512 + n] : 0.f;
  Wt[idx] = (_Float16)v;
}

// ---------------------------------------------------------------------------
// Embedding: one-hot concat of 7 categorical feats (vocab 45,8,5,6,2,2,8),
// L2-normalized (7 ones per row -> every nonzero = 1/sqrt(7)). f16, K-padded.
// ---------------------------------------------------------------------------
__global__ void embed_kernel(const int* __restrict__ x, _Float16* __restrict__ h0) {
  int idx = blockIdx.x * blockDim.x + threadIdx.x;
  if (idx >= MP_NODES * K0_PAD) return;
  int n = idx / K0_PAD, k = idx - n * K0_PAD;
  float v = 0.f;
  if (n < NN_NODES && k < 76) {
    const int off[7] = {0, 45, 53, 58, 64, 66, 68};
    const int sz[7]  = {45, 8, 5, 6, 2, 2, 8};
#pragma unroll
    for (int i = 0; i < 7; ++i)
      if (k >= off[i] && k < off[i] + sz[i] && x[n * 7 + i] == (k - off[i]))
        v = 0.3779644730092272f;   // 1/sqrt(7)
  }
  h0[idx] = (_Float16)v;
}

__global__ void zero_pad_h16_kernel(_Float16* __restrict__ h16) {
  int idx = blockIdx.x * blockDim.x + threadIdx.x;
  if (idx < (MP_NODES - NN_NODES) * HC)
    h16[(size_t)NN_NODES * HC + idx] = (_Float16)0.f;
}

__global__ void zero_f32_kernel(float* __restrict__ p, int n) {
  int i = blockIdx.x * blockDim.x + threadIdx.x;
  if (i < n) p[i] = 0.f;
}

// ---------------------------------------------------------------------------
// WMMA GEMM: C[Mpad x N] = A_f16[Mpad x K] * Bt_f16[N x K]^T + bias.
// Block = 256 threads (8 waves); wave -> 16 rows x 64 cols (4 accumulators).
// A layout: lane<16 holds K {b..b+15} with b=0, lane>=16
// with b=16. B (from transposed weights): lane holds col (lane&15), 16
// contiguous K values starting at (lane&16 ? 16 : 0).
// ---------------------------------------------------------------------------
__global__ void __launch_bounds__(256) gemm_f16_wmma_kernel(
    const _Float16* __restrict__ A, const _Float16* __restrict__ Bt,
    const float* __restrict__ bias, float* __restrict__ C,
    int Mpad, int K, int N) {
  const int lane = threadIdx.x & 31;
  const int wave = threadIdx.x >> 5;
  const int m0 = (blockIdx.x * 8 + wave) * 16;
  if (m0 >= Mpad) return;                    // wave-uniform: EXEC stays all-1s
  const int n0 = blockIdx.y * 64;

  const int arow = m0 + (lane & 15);
  const int ab   = (lane & 16) ? 16 : 0;
  const _Float16* Ap = A + (size_t)arow * K + ab;

  const int bk = (lane & 16) ? 16 : 0;
  const int nc = n0 + (lane & 15);
  const _Float16* Bp[4];
#pragma unroll
  for (int t = 0; t < 4; ++t) Bp[t] = Bt + (size_t)(nc + t * 16) * K + bk;

  v8f acc[4] = {v8f{}, v8f{}, v8f{}, v8f{}};
  for (int k = 0; k < K; k += 32) {
    v8h alo = *(const v8h*)(Ap + k);
    v8h ahi = *(const v8h*)(Ap + k + 8);
    v16h a;
#pragma unroll
    for (int i = 0; i < 8; ++i) { a[i] = alo[i]; a[8 + i] = ahi[i]; }
#pragma unroll
    for (int t = 0; t < 4; ++t) {
      v16h b = *(const v16h*)(Bp[t] + k);
      acc[t] = __builtin_amdgcn_wmma_f32_16x16x32_f16(
          /*neg_a=*/false, a, /*neg_b=*/false, b,
          /*c_mod=*/(short)0, acc[t], /*reuse_a=*/false, /*reuse_b=*/false);
    }
  }
  // C/D layout: VGPR r -> row r (lanes 0-15) / row 8+r (lanes 16-31), col lane&15
  const int crow = m0 + ((lane & 16) ? 8 : 0);
#pragma unroll
  for (int t = 0; t < 4; ++t) {
    const int cc = n0 + t * 16 + (lane & 15);
    const float bv = bias[cc];
#pragma unroll
    for (int r = 0; r < 8; ++r)
      C[(size_t)(crow + r) * N + cc] = acc[t][r] + bv;
  }
}

// ---------------------------------------------------------------------------
// Per-layer init: num=0, den=0, amax=-inf(mapped), BN sum/sumsq=0.
// ---------------------------------------------------------------------------
__global__ void layer_init_kernel(float* __restrict__ num, float* __restrict__ den,
                                  unsigned* __restrict__ amax, float* __restrict__ stats,
                                  int total, int mp4) {
  int i = blockIdx.x * blockDim.x + threadIdx.x;
  if (i < total) num[i] = 0.f;
  if (i < mp4) { den[i] = 0.f; amax[i] = AMAX_NEG_INF; }
  if (i < 2 * HC) stats[i] = 0.f;
}

// ---------------------------------------------------------------------------
// Edge pass 1: alpha[e][h] = <q[dst], k[src]+E(e)>_h / sqrt(C); atomicMax amax.
// One wave per edge; lane -> 16 contiguous channels; 8 lanes per head.
// ---------------------------------------------------------------------------
__global__ void __launch_bounds__(256) edge_pass1_kernel(
    const int* __restrict__ ei, const float* __restrict__ ea,
    const float* __restrict__ Q, const float* __restrict__ Km,
    const float* __restrict__ We, const float* __restrict__ be,
    float* __restrict__ alpha, unsigned* __restrict__ amax) {
  int gw = blockIdx.x * 8 + (threadIdx.x >> 5);
  int lane = threadIdx.x & 31;
  if (gw >= N_EDGES) return;
  int src = ei[gw];
  int dst = ei[N_EDGES + gw];
  float e0 = ea[2 * gw], e1 = ea[2 * gw + 1];
  const float* qrow = Q + ((size_t)dst << 9);
  const float* krow = Km + ((size_t)src << 9);
  int c0 = lane * 16;
  float p = 0.f;
#pragma unroll
  for (int i = 0; i < 16; ++i) {
    int c = c0 + i;
    float kj = krow[c] + e0 * We[c] + e1 * We[512 + c] + be[c];
    p += qrow[c] * kj;
  }
  p += __shfl_xor(p, 1, 32);
  p += __shfl_xor(p, 2, 32);
  p += __shfl_xor(p, 4, 32);
  if ((lane & 7) == 0) {
    int h = lane >> 3;
    float a = p * 0.08838834764831845f;   // 1/sqrt(128)
    alpha[(size_t)gw * 4 + h] = a;
    atomicMax(&amax[(size_t)dst * 4 + h], fmap(a));
  }
}

// ---------------------------------------------------------------------------
// Edge pass 2: ex = exp(alpha - amax[dst]); den += ex; num += ex*(v[src]+E(e)).
// out = num/den later, so softmax needs only two edge passes.
// ---------------------------------------------------------------------------
__global__ void __launch_bounds__(256) edge_pass2_kernel(
    const int* __restrict__ ei, const float* __restrict__ ea,
    const float* __restrict__ V, const float* __restrict__ We,
    const float* __restrict__ be, const float* __restrict__ alpha,
    const unsigned* __restrict__ amax, float* __restrict__ den,
    float* __restrict__ num) {
  int gw = blockIdx.x * 8 + (threadIdx.x >> 5);
  int lane = threadIdx.x & 31;
  if (gw >= N_EDGES) return;
  int src = ei[gw];
  int dst = ei[N_EDGES + gw];
  float e0 = ea[2 * gw], e1 = ea[2 * gw + 1];
  int h = lane >> 3;
  float a  = alpha[(size_t)gw * 4 + h];
  float mx = funmap(amax[(size_t)dst * 4 + h]);
  float ex = __expf(a - mx);
  if ((lane & 7) == 0) atomicAdd(&den[(size_t)dst * 4 + h], ex);
  const float* vrow = V + ((size_t)src << 9);
  float* nrow = num + ((size_t)dst << 9);
  int c0 = lane * 16;
#pragma unroll
  for (int i = 0; i < 16; ++i) {
    int c = c0 + i;
    float vj = vrow[c] + e0 * We[c] + e1 * We[512 + c] + be[c];
    atomicAdd(&nrow[c], ex * vj);
  }
}

// ---------------------------------------------------------------------------
// Node epilogue: t = num/(den+1e-16) + skip; in-place into num; BN partial
// sums (block covers 64 nodes x all 512 channels, thread = channel).
// ---------------------------------------------------------------------------
#define EPI_CHUNK 64
__global__ void __launch_bounds__(512) epilogue_kernel(
    const float* __restrict__ S, const float* __restrict__ den,
    float* __restrict__ num, float* __restrict__ stats) {
  int c = threadIdx.x;
  int h = c >> 7;
  int n0 = blockIdx.x * EPI_CHUNK;
  int n1 = n0 + EPI_CHUNK; if (n1 > NN_NODES) n1 = NN_NODES;
  float s = 0.f, sq = 0.f;
  for (int n = n0; n < n1; ++n) {
    size_t i = ((size_t)n << 9) + c;
    float d = den[(size_t)n * 4 + h];
    float t = num[i] / (d + 1e-16f) + S[i];
    num[i] = t;
    s += t; sq += t * t;
  }
  atomicAdd(&stats[c], s);
  atomicAdd(&stats[HC + c], sq);
}

__global__ void bn_finalize_kernel(float* __restrict__ stats, float invN) {
  int c = blockIdx.x * blockDim.x + threadIdx.x;
  if (c >= HC) return;
  float m = stats[c] * invN;
  float v = stats[HC + c] * invN - m * m;
  stats[2 * HC + c] = m;
  stats[3 * HC + c] = rsqrtf(v + 1e-5f);
}

// BN + LeakyReLU(0.01); write next-layer f16 activations and f32 copy (pool).
__global__ void normalize_kernel(const float* __restrict__ t, const float* __restrict__ stats,
                                 const float* __restrict__ g, const float* __restrict__ b,
                                 _Float16* __restrict__ h16, float* __restrict__ hf32) {
  int idx = blockIdx.x * blockDim.x + threadIdx.x;
  if (idx >= NN_NODES * HC) return;
  int c = idx & 511;
  float y = (t[idx] - stats[2 * HC + c]) * stats[3 * HC + c] * g[c] + b[c];
  y = (y >= 0.f) ? y : 0.01f * y;
  h16[idx]  = (_Float16)y;
  hf32[idx] = y;
}

// ---------------------------------------------------------------------------
// global_add_pool: pool[batch[n]][c] += h[n][c]
// ---------------------------------------------------------------------------
__global__ void pool_kernel(const float* __restrict__ h, const int* __restrict__ batch,
                            float* __restrict__ pool) {
  int idx = blockIdx.x * blockDim.x + threadIdx.x;
  if (idx >= NN_NODES * HC) return;
  int n = idx >> 9, c = idx & 511;
  atomicAdd(pool + ((size_t)batch[n] << 9) + c, h[idx]);
}

// ---------------------------------------------------------------------------
// Small MLP pieces (512x{512,64,64} GEMMs are tiny -> plain kernels)
// ---------------------------------------------------------------------------
__global__ void mlp_gemm_kernel(const float* __restrict__ A, const float* __restrict__ W,
                                const float* __restrict__ b, float* __restrict__ out,
                                int G, int K, int N) {
  int idx = blockIdx.x * blockDim.x + threadIdx.x;
  if (idx >= G * N) return;
  int g = idx / N, j = idx - g * N;
  float s = b[j];
  for (int k = 0; k < K; ++k) s += A[(size_t)g * K + k] * W[(size_t)k * N + j];
  out[idx] = s;
}

__global__ void bn_small_stats_kernel(const float* __restrict__ X, float* __restrict__ mrs,
                                      int G, int N) {
  int j = threadIdx.x;
  if (j >= N) return;
  float s = 0.f, sq = 0.f;
  for (int g = 0; g < G; ++g) { float v = X[(size_t)g * N + j]; s += v; sq += v * v; }
  float m = s / G;
  float var = sq / G - m * m;
  mrs[j] = m;
  mrs[N + j] = rsqrtf(var + 1e-5f);
}

__global__ void bn_relu_kernel(float* __restrict__ X, const float* __restrict__ mrs,
                               const float* __restrict__ g, const float* __restrict__ b,
                               int G, int N) {
  int idx = blockIdx.x * blockDim.x + threadIdx.x;
  if (idx >= G * N) return;
  int j = idx % N;
  float y = (X[idx] - mrs[j]) * mrs[N + j] * g[j] + b[j];
  X[idx] = (y > 0.f) ? y : 0.f;
}

// ---------------------------------------------------------------------------
// Host launcher
// ---------------------------------------------------------------------------
extern "C" void kernel_launch(void* const* d_in, const int* in_sizes, int n_in,
                              void* d_out, int out_size, void* d_ws, size_t ws_size,
                              hipStream_t stream) {
  (void)in_sizes; (void)n_in; (void)out_size; (void)ws_size;
  const int* x     = (const int*)d_in[0];
  const int* ei    = (const int*)d_in[1];
  const float* ea  = (const float*)d_in[2];
  const int* batch = (const int*)d_in[3];

  // params in JAX pytree (sorted-key) flatten order:
  // per layer: 0 bn_b, 1 bn_g, 2 e.b, 3 e.w, 4 k.b, 5 k.w, 6 q.b, 7 q.w,
  //            8 skip.b, 9 skip.w, 10 v.b, 11 v.w
  const float* Lp[4][12];
  for (int l = 0; l < 4; ++l)
    for (int j = 0; j < 12; ++j) Lp[l][j] = (const float*)d_in[4 + l * 12 + j];
  const float* bn1_b = (const float*)d_in[52];
  const float* bn1_g = (const float*)d_in[53];
  const float* bn2_b = (const float*)d_in[54];
  const float* bn2_g = (const float*)d_in[55];
  const float* l1_b  = (const float*)d_in[56];
  const float* l1_w  = (const float*)d_in[57];
  const float* l2_b  = (const float*)d_in[58];
  const float* l2_w  = (const float*)d_in[59];
  const float* l3_b  = (const float*)d_in[60];
  const float* l3_w  = (const float*)d_in[61];

  // ---- workspace carve-up (256B aligned regions) ----
  char* ws = (char*)d_ws;
  size_t off = 0;
  auto alloc = [&](size_t bytes) -> void* {
    void* p = ws + off;
    off = (off + bytes + 255) & ~(size_t)255;
    return p;
  };
  _Float16* wT[4][4];                       // [layer][q,k,v,skip]
  for (int l = 0; l < 4; ++l) {
    int Kp = (l == 0) ? K0_PAD : HC;
    for (int m = 0; m < 4; ++m) wT[l][m] = (_Float16*)alloc((size_t)512 * Kp * 2);
  }
  _Float16* h0  = (_Float16*)alloc((size_t)MP_NODES * K0_PAD * 2);
  _Float16* h16 = (_Float16*)alloc((size_t)MP_NODES * HC * 2);
  float* Qb  = (float*)alloc((size_t)MP_NODES * HC * 4);
  float* Kb  = (float*)alloc((size_t)MP_NODES * HC * 4);
  float* Vb  = (float*)alloc((size_t)MP_NODES * HC * 4);
  float* Sb  = (float*)alloc((size_t)MP_NODES * HC * 4);   // skip; reused as f32 h
  float* num = (float*)alloc((size_t)MP_NODES * HC * 4);   // reused as t (pre-BN)
  float* den = (float*)alloc((size_t)MP_NODES * 4 * 4);
  unsigned* amax = (unsigned*)alloc((size_t)MP_NODES * 4 * 4);
  float* alpha = (float*)alloc((size_t)N_EDGES * 4 * 4);
  float* stats = (float*)alloc((size_t)4 * HC * 4);        // sum, sumsq, mean, rs
  float* pool  = (float*)alloc((size_t)N_GRAPHS * HC * 4);
  float* g1    = (float*)alloc((size_t)N_GRAPHS * 64 * 4);
  float* g2    = (float*)alloc((size_t)N_GRAPHS * 64 * 4);
  float* mrs1  = (float*)alloc(128 * 4);
  float* mrs2  = (float*)alloc(128 * 4);

  const int wmap[4] = {7, 5, 11, 9};        // q.w, k.w, v.w, skip.w
  const int bmap[4] = {6, 4, 10, 8};        // q.b, k.b, v.b, skip.b

  // ---- weight prep (f32 -> transposed f16, K zero-padded) ----
  for (int l = 0; l < 4; ++l) {
    int fin = (l == 0) ? 76 : HC;
    int Kp  = (l == 0) ? K0_PAD : HC;
    int nthr = 512 * Kp;
    for (int m = 0; m < 4; ++m)
      wprep_kernel<<<(nthr + 255) / 256, 256, 0, stream>>>(Lp[l][wmap[m]], wT[l][m], fin, Kp);
  }

  // ---- embedding + pad-row zeroing ----
  embed_kernel<<<(MP_NODES * K0_PAD + 255) / 256, 256, 0, stream>>>(x, h0);
  zero_pad_h16_kernel<<<((MP_NODES - NN_NODES) * HC + 255) / 256, 256, 0, stream>>>(h16);

  const dim3 gemm_grid((MP_NODES / 16 + 7) / 8, HC / 64);
  const int elem_blocks = (NN_NODES * HC + 255) / 256;
  const int edge_blocks = (N_EDGES + 7) / 8;

  // ---- transformer-conv layers ----
  for (int l = 0; l < 4; ++l) {
    const _Float16* A = (l == 0) ? h0 : h16;
    const int K = (l == 0) ? K0_PAD : HC;

    layer_init_kernel<<<(MP_NODES * HC + 255) / 256, 256, 0, stream>>>(
        num, den, amax, stats, MP_NODES * HC, MP_NODES * 4);

    float* outs[4] = {Qb, Kb, Vb, Sb};
    for (int m = 0; m < 4; ++m)
      gemm_f16_wmma_kernel<<<gemm_grid, 256, 0, stream>>>(
          A, wT[l][m], Lp[l][bmap[m]], outs[m], MP_NODES, K, HC);

    edge_pass1_kernel<<<edge_blocks, 256, 0, stream>>>(
        ei, ea, Qb, Kb, Lp[l][3], Lp[l][2], alpha, amax);
    edge_pass2_kernel<<<edge_blocks, 256, 0, stream>>>(
        ei, ea, Vb, Lp[l][3], Lp[l][2], alpha, amax, den, num);

    epilogue_kernel<<<(NN_NODES + EPI_CHUNK - 1) / EPI_CHUNK, 512, 0, stream>>>(
        Sb, den, num, stats);
    bn_finalize_kernel<<<2, 256, 0, stream>>>(stats, 1.0f / NN_NODES);
    normalize_kernel<<<elem_blocks, 256, 0, stream>>>(
        num, stats, Lp[l][1], Lp[l][0], h16, Sb);
  }

  // ---- global add-pool ----
  zero_f32_kernel<<<(N_GRAPHS * HC + 255) / 256, 256, 0, stream>>>(pool, N_GRAPHS * HC);
  pool_kernel<<<elem_blocks, 256, 0, stream>>>(Sb, batch, pool);

  // ---- MLP head: 512 -> 64 -> 64 -> 2 with BN+ReLU ----
  mlp_gemm_kernel<<<(N_GRAPHS * 64 + 255) / 256, 256, 0, stream>>>(
      pool, l1_w, l1_b, g1, N_GRAPHS, HC, 64);
  bn_small_stats_kernel<<<1, 64, 0, stream>>>(g1, mrs1, N_GRAPHS, 64);
  bn_relu_kernel<<<(N_GRAPHS * 64 + 255) / 256, 256, 0, stream>>>(
      g1, mrs1, bn1_g, bn1_b, N_GRAPHS, 64);

  mlp_gemm_kernel<<<(N_GRAPHS * 64 + 255) / 256, 256, 0, stream>>>(
      g1, l2_w, l2_b, g2, N_GRAPHS, 64, 64);
  bn_small_stats_kernel<<<1, 64, 0, stream>>>(g2, mrs2, N_GRAPHS, 64);
  bn_relu_kernel<<<(N_GRAPHS * 64 + 255) / 256, 256, 0, stream>>>(
      g2, mrs2, bn2_g, bn2_b, N_GRAPHS, 64);

  mlp_gemm_kernel<<<(N_GRAPHS * 2 + 255) / 256, 256, 0, stream>>>(
      g2, l3_w, l3_b, (float*)d_out, N_GRAPHS, 64, 2);
}
